// PSSModel_27127013441796
// MI455X (gfx1250) — matrix operand order repaired and checked by reading, not compiled, not run on hardware
//
#include <hip/hip_runtime.h>
#include <math.h>

// ---------------------------------------------------------------------------
// GVP-GNN forward for MI455X (gfx1250, wave32).
// All dense matmuls go through one generic bf16-WMMA GEMM kernel
// (v_wmma_f32_16x16x32_bf16, fp32 accumulate). LDS tiles are stored in WMMA
// fragment order; fragments are fetched as contiguous ds_load_b128 pairs and
// staged with wide ds_store_b128/b64. The k-loop is split into an unguarded
// full-tile main loop plus one guarded tail iteration.
// ---------------------------------------------------------------------------

#define EPSV 1e-8f

constexpr int NNODES = 4096;
constexpr int NEDGES = 131072;
constexpr int NAA    = 21;
constexpr int NS     = 200;   // 2*VS_H
constexpr int ESCAT  = 53;    // ES_H + AA

typedef __bf16 bf16_t;
typedef __attribute__((ext_vector_type(16))) __bf16 v16bf;
typedef __attribute__((ext_vector_type(8)))  __bf16 v8bf;
typedef __attribute__((ext_vector_type(4)))  __bf16 v4bf;
typedef __attribute__((ext_vector_type(8)))  float  v8f;

// ---------------------------------------------------------------------------
// Generic GEMM:  C[m,n(,z)] = act( sum_k A[m,k,z] * B[k,n] + bias[n] (+ C[m,n(,z)]) )
//   A element:  base[(row(m))*a_ld + k*a_ks + z*a_zs]
//     GATHER: row(m)=rowidx[m];  MASK: base = mask[m] ? A0 : A1;  CONTIG: a_ks==1
//   B element:  B[k*b_ks + n*b_ns]        (b_ks=1,b_ns=ld gives B^T)
//   C element:  C[m*c_ld + n*c_ns + z*c_zs]
// act: 0 none, 1 relu, 2 sigmoid.  accumulate: old C is added before act.
// Block = 8 waves; each wave computes a 16x32 slab (2 WMMA tiles sharing A).
// ---------------------------------------------------------------------------
template<bool GATHER, bool MASK, bool CONTIG>
__global__ __launch_bounds__(256)
void k_gemm(const float* __restrict__ A0, const float* __restrict__ A1,
            const int* __restrict__ rowidx, const int* __restrict__ mask,
            const float* __restrict__ B, const float* __restrict__ bias,
            float* __restrict__ C,
            int M, int N, int K,
            int a_ld, int a_ks, int a_zs,
            int b_ks, int b_ns,
            int c_ld, int c_ns, int c_zs,
            int accumulate, int act)
{
    __shared__ __align__(32) bf16_t sA[8][32][16];   // [mtile][lane][elem]
    __shared__ __align__(32) bf16_t sB[2][32][16];   // [ntile][lane][elem]
    const int tid  = threadIdx.x;
    const int lane = tid & 31;
    const int wave = tid >> 5;
    const int hlf  = lane >> 4;
    const int l16  = lane & 15;
    const int z    = blockIdx.z;
    const int mB   = blockIdx.y * 128;
    const int n0   = blockIdx.x * 32;

    // --- A staging: one row per thread, 16 consecutive k (two 8-wide runs) ---
    const int arow = tid >> 1;           // 0..127
    const int ak0  = (tid & 1) << 4;     // 0 or 16
    const int am   = mB + arow;
    const float* arp = nullptr;
    if (am < M) {
        int r = GATHER ? rowidx[am] : am;
        const float* base = A0;
        if (MASK) base = mask[am] ? A0 : A1;
        arp = base + (long)r * a_ld + (long)z * a_zs;
    }
    const int smt = arow >> 4;
    const int sml = arow & 15;

    // --- B staging: one n per thread, 4 consecutive k ---
    const int bn  = tid & 31;
    const int bk0 = (tid >> 5) << 2;     // 0,4,...,28
    const int bnn = n0 + bn;
    bf16_t* bdst = &sB[bn >> 4][(((bk0 >> 3) & 1) << 4) + (bn & 15)][((bk0 >> 4) << 3) + (bk0 & 7)];

    v8f acc0 = {0.f,0.f,0.f,0.f,0.f,0.f,0.f,0.f};
    v8f acc1 = {0.f,0.f,0.f,0.f,0.f,0.f,0.f,0.f};

    const int Kmain = K & ~31;
    int kt = 0;
    // ---------------- main loop: full 32-wide k tiles, unguarded ----------------
    for (; kt < Kmain; kt += 32) {
        if (arp && kt + 64 <= K)
            __builtin_prefetch(arp + (long)(kt + 32) * (CONTIG ? 1 : a_ks), 0, 1);
        #pragma unroll
        for (int run = 0; run < 2; ++run) {
            const int kk0 = ak0 + (run << 3);
            v8bf pk;
            if (arp) {
                #pragma unroll
                for (int w = 0; w < 8; ++w)
                    pk[w] = (bf16_t)(CONTIG ? arp[kt + kk0 + w]
                                            : arp[(long)(kt + kk0 + w) * a_ks]);
            } else {
                #pragma unroll
                for (int w = 0; w < 8; ++w) pk[w] = (bf16_t)0.f;
            }
            const int g = kk0 >> 4;
            const int h = (kk0 >> 3) & 1;
            *(v8bf*)&sA[smt][(h << 4) + sml][g << 3] = pk;   // ds_store_b128
        }
        {
            v4bf pk;
            if (bnn < N) {
                #pragma unroll
                for (int j = 0; j < 4; ++j)
                    pk[j] = (bf16_t)B[(long)(kt + bk0 + j) * b_ks + (long)bnn * b_ns];
            } else {
                #pragma unroll
                for (int j = 0; j < 4; ++j) pk[j] = (bf16_t)0.f;
            }
            *(v4bf*)bdst = pk;                               // ds_store_b64
        }
        __syncthreads();
        v16bf af = *(const v16bf*)&sA[wave][lane][0];
        v16bf b0 = *(const v16bf*)&sB[0][lane][0];
        v16bf b1 = *(const v16bf*)&sB[1][lane][0];
        acc0 = __builtin_amdgcn_wmma_f32_16x16x32_bf16(false, af, false, b0, (short)0, acc0, false, false);
        acc1 = __builtin_amdgcn_wmma_f32_16x16x32_bf16(false, af, false, b1, (short)0, acc1, false, false);
        __syncthreads();
    }
    // ---------------- tail: K%32 remainder, guarded per element ----------------
    if (kt < K) {
        #pragma unroll
        for (int w = 0; w < 16; ++w) {
            int kk = ak0 + w;
            int k  = kt + kk;
            float v = 0.f;
            if (arp && k < K) v = CONTIG ? arp[k] : arp[(long)k * a_ks];
            int g = kk >> 4;
            int h = (kk >> 3) & 1;
            int e = (g << 3) + (kk & 7);
            sA[smt][(h << 4) + sml][e] = (bf16_t)v;
        }
        {
            v4bf pk;
            #pragma unroll
            for (int j = 0; j < 4; ++j) {
                int k = kt + bk0 + j;
                float v = 0.f;
                if (k < K && bnn < N) v = B[(long)k * b_ks + (long)bnn * b_ns];
                pk[j] = (bf16_t)v;
            }
            *(v4bf*)bdst = pk;
        }
        __syncthreads();
        v16bf af = *(const v16bf*)&sA[wave][lane][0];
        v16bf b0 = *(const v16bf*)&sB[0][lane][0];
        v16bf b1 = *(const v16bf*)&sB[1][lane][0];
        acc0 = __builtin_amdgcn_wmma_f32_16x16x32_bf16(false, af, false, b0, (short)0, acc0, false, false);
        acc1 = __builtin_amdgcn_wmma_f32_16x16x32_bf16(false, af, false, b1, (short)0, acc1, false, false);
    }
    // ---- epilogue: C/D layout -> vgpr r: m = base + half*8 + r, n = l16 ----
    int mw = mB + wave * 16;
    #pragma unroll
    for (int r = 0; r < 8; ++r) {
        int m = mw + hlf * 8 + r;
        if (m >= M) continue;
        int na = n0 + l16;
        if (na < N) {
            long off = (long)m * c_ld + (long)na * c_ns + (long)z * c_zs;
            float v = acc0[r];
            if (bias) v += bias[na];
            if (accumulate) v += C[off];
            if (act == 1)      v = fmaxf(v, 0.f);
            else if (act == 2) v = 1.f / (1.f + __expf(-v));
            C[off] = v;
        }
        int nb = n0 + 16 + l16;
        if (nb < N) {
            long off = (long)m * c_ld + (long)nb * c_ns + (long)z * c_zs;
            float v = acc1[r];
            if (bias) v += bias[nb];
            if (accumulate) v += C[off];
            if (act == 1)      v = fmaxf(v, 0.f);
            else if (act == 2) v = 1.f / (1.f + __expf(-v));
            C[off] = v;
        }
    }
}

// ------------------------- elementwise / reduction -------------------------
__global__ void k_fill(float* __restrict__ p, long n) {
    long i = blockIdx.x * (long)blockDim.x + threadIdx.x;
    if (i < n) p[i] = 0.f;
}

// vn[m,h] = sqrt(max(sum_c v[m,h,c]^2, eps))
__global__ void k_vnorm(const float* __restrict__ v, float* __restrict__ o,
                        long M, int H, int vld, int old_) {
    long i = blockIdx.x * (long)blockDim.x + threadIdx.x;
    if (i >= M * (long)H) return;
    long m = i / H; int h = (int)(i % H);
    long b = m * (long)vld + (long)h * 3;
    float x = v[b], y = v[b+1], z = v[b+2];
    o[m * (long)old_ + h] = sqrtf(fmaxf(x*x + y*y + z*z, EPSV));
}

// v *= sigmoid(||v||) per channel
__global__ void k_vgate(float* __restrict__ v, long M, int H, int ld) {
    long i = blockIdx.x * (long)blockDim.x + threadIdx.x;
    if (i >= M * (long)H) return;
    long m = i / H; int h = (int)(i % H);
    long b = m * (long)ld + (long)h * 3;
    float x = v[b], y = v[b+1], z = v[b+2];
    float n = sqrtf(fmaxf(x*x + y*y + z*z, EPSV));
    float s = 1.f / (1.f + __expf(-n));
    v[b] = x*s; v[b+1] = y*s; v[b+2] = z*s;
}

// in-place LayerNorm over D (<=256) columns of each row (row = blockIdx.x)
__global__ __launch_bounds__(256)
void k_layernorm(float* __restrict__ x, const float* __restrict__ g,
                 const float* __restrict__ b, int D, int ld) {
    __shared__ float s1[256], s2[256];
    long row = blockIdx.x;
    int t = threadIdx.x;
    float v = (t < D) ? x[row * (long)ld + t] : 0.f;
    s1[t] = v; s2[t] = v * v;
    __syncthreads();
    for (int s = 128; s > 0; s >>= 1) {
        if (t < s) { s1[t] += s1[t+s]; s2[t] += s2[t+s]; }
        __syncthreads();
    }
    float mu  = s1[0] / D;
    float var = s2[0] / D - mu * mu;
    float inv = rsqrtf(var + 1e-5f);
    if (t < D) x[row * (long)ld + t] = (v - mu) * inv * g[t] + b[t];
}

// GVP vector LN: v *= rsqrt(mean_h(max(||v_h||^2, eps)))  (H<=32, block=32)
__global__ __launch_bounds__(32)
void k_vscale(float* __restrict__ v, int H, int ld) {
    __shared__ float sm[32];
    long row = blockIdx.x; int t = threadIdx.x;
    float a = 0.f, bb = 0.f, c = 0.f, n = 0.f;
    if (t < H) {
        long b = row * (long)ld + (long)t * 3;
        a = v[b]; bb = v[b+1]; c = v[b+2];
        n = fmaxf(a*a + bb*bb + c*c, EPSV);
    }
    sm[t] = n; __syncthreads();
    for (int s = 16; s > 0; s >>= 1) { if (t < s) sm[t] += sm[t+s]; __syncthreads(); }
    float sc = rsqrtf(sm[0] / H);
    if (t < H) {
        long b = row * (long)ld + (long)t * 3;
        v[b] = a*sc; v[b+1] = bb*sc; v[b+2] = c*sc;
    }
}

__global__ void k_scatter_add(const float* __restrict__ s, int s_ld,
                              const int* __restrict__ idx,
                              float* __restrict__ o, int o_ld, long E, int D) {
    long i = blockIdx.x * (long)blockDim.x + threadIdx.x;
    if (i >= E * (long)D) return;
    long e = i / D; int d = (int)(i % D);
    atomicAdd(&o[(long)idx[e] * o_ld + d], s[e * (long)s_ld + d]);
}

__global__ void k_scatter_deg(const int* __restrict__ idx, float* __restrict__ deg, long E) {
    long e = blockIdx.x * (long)blockDim.x + threadIdx.x;
    if (e < E) atomicAdd(&deg[idx[e]], 1.f);
}

__global__ void k_residual(float* __restrict__ x, int x_ld,
                           const float* __restrict__ dh, int d_ld,
                           const float* __restrict__ deg, long R, int D) {
    long i = blockIdx.x * (long)blockDim.x + threadIdx.x;
    if (i >= R * (long)D) return;
    long r = i / D; int d = (int)(i % D);
    float val = dh[r * (long)d_ld + d];
    if (deg) val /= fmaxf(deg[r], 1.f);
    x[r * (long)x_ld + d] += val;
}

__global__ void k_eemb(float* __restrict__ ecat, const int* __restrict__ src,
                       const int* __restrict__ dst, const int* __restrict__ seq,
                       const float* __restrict__ emb, long E) {
    long i = blockIdx.x * (long)blockDim.x + threadIdx.x;
    if (i >= E * (long)NAA) return;
    long e = i / NAA; int j = (int)(i % NAA);
    float v = (src[e] < dst[e]) ? emb[(long)seq[src[e]] * NAA + j] : 0.f;
    ecat[e * (long)ESCAT + 32 + j] = v;
}

__global__ void k_armask(const int* __restrict__ src, const int* __restrict__ dst,
                         int* __restrict__ m, long E) {
    long e = blockIdx.x * (long)blockDim.x + threadIdx.x;
    if (e < E) m[e] = (src[e] < dst[e]) ? 1 : 0;
}

// softmax over D columns of row blockIdx.x, with pre-scale
__global__ __launch_bounds__(256)
void k_softmax(float* __restrict__ a, int D, float scale) {
    __shared__ float sm[256];
    long row = blockIdx.x; int t = threadIdx.x;
    float mx = -1e30f;
    for (int j = t; j < D; j += 256) mx = fmaxf(mx, a[row * (long)D + j] * scale);
    sm[t] = mx; __syncthreads();
    for (int s = 128; s > 0; s >>= 1) { if (t < s) sm[t] = fmaxf(sm[t], sm[t+s]); __syncthreads(); }
    mx = sm[0]; __syncthreads();
    float sum = 0.f;
    for (int j = t; j < D; j += 256) {
        float e = __expf(a[row * (long)D + j] * scale - mx);
        a[row * (long)D + j] = e; sum += e;
    }
    sm[t] = sum; __syncthreads();
    for (int s = 128; s > 0; s >>= 1) { if (t < s) sm[t] += sm[t+s]; __syncthreads(); }
    float inv = 1.f / sm[0];
    for (int j = t; j < D; j += 256) a[row * (long)D + j] *= inv;
}

// sequential GRU scan: blockDim == H, one block. xg = x@Wih+bih precomputed.
__global__ void k_gru_scan(const float* __restrict__ xg, const float* __restrict__ whh,
                           const float* __restrict__ bhh, float* __restrict__ out,
                           int T, int H, int reverse, int out_ld, int out_off) {
    __shared__ float hbuf[256];
    int j = threadIdx.x;
    hbuf[j] = 0.f; __syncthreads();
    for (int t = 0; t < T; ++t) {
        int idx = reverse ? (T - 1 - t) : t;
        const float* xt = xg + (long)idx * 3 * H;
        float ghr = bhh[j], ghz = bhh[H + j], ghn = bhh[2*H + j];
        for (int i = 0; i < H; ++i) {
            float hi = hbuf[i];
            const float* wr = whh + (long)i * 3 * H;
            ghr += hi * wr[j];
            ghz += hi * wr[H + j];
            ghn += hi * wr[2*H + j];
        }
        float r  = 1.f / (1.f + __expf(-(xt[j]       + ghr)));
        float zz = 1.f / (1.f + __expf(-(xt[H + j]   + ghz)));
        float ng = tanhf(xt[2*H + j] + r * ghn);
        float hn = (1.f - zz) * ng + zz * hbuf[j];
        __syncthreads();
        hbuf[j] = hn;
        out[(long)idx * out_ld + out_off + j] = hn;
        __syncthreads();
    }
}

// ------------------------------ host side ----------------------------------
struct GvpP { const float *wh, *ws, *bs, *wv; };
struct LnP  { const float *g, *b; };
struct GruP { const float *wih, *whh, *bih, *bhh; };
struct ConvP { GvpP m0, m1, m2, f0, f1; LnP n0, n1; };

extern "C" void kernel_launch(void* const* d_in, const int* in_sizes, int n_in,
                              void* d_out, int out_size, void* d_ws, size_t ws_size,
                              hipStream_t stream)
{
    (void)in_sizes; (void)out_size; (void)ws_size;
    const long N = NNODES, E = NEDGES;

    const int*   seq       = (const int*)  d_in[0];
    const float* node_s    = (const float*)d_in[1];
    const float* node_v    = (const float*)d_in[2];
    const float* node_no_s = (const float*)d_in[3];
    const float* node_no_v = (const float*)d_in[4];
    const float* edge_s    = (const float*)d_in[5];
    const float* edge_v    = (const float*)d_in[6];
    const int*   eidx      = (const int*)  d_in[7];
    const int*   srcI = eidx;
    const int*   dstI = eidx + NEDGES;

    // param leaves in jax tree-flatten (alphabetical) order, after the 8 arrays
    auto P = [&](int leaf) -> const float* {
        int i = 8 + leaf;
        if (i >= n_in) i = (n_in > 0 ? n_in - 1 : 0);
        return (const float*)d_in[i];
    };
    auto convp_at = [&](int b) -> ConvP {
        ConvP c;
        c.f0 = {P(b+1),  P(b+2),  P(b+0),  P(b+3)};
        c.f1 = {P(b+5),  P(b+6),  P(b+4),  P(b+7)};
        c.m0 = {P(b+9),  P(b+10), P(b+8),  P(b+11)};
        c.m1 = {P(b+13), P(b+14), P(b+12), P(b+15)};
        c.m2 = {P(b+17), P(b+18), P(b+16), P(b+19)};
        c.n0 = {P(b+21), P(b+20)};
        c.n1 = {P(b+23), P(b+22)};
        return c;
    };
    auto grup_at = [&](int b) -> GruP { return {P(b+3), P(b+2), P(b+1), P(b+0)}; };

    // workspace bump allocator
    char* wsb = (char*)d_ws; size_t woff = 0;
    auto alloc = [&](size_t nf) -> float* {
        float* p = (float*)(wsb + woff);
        woff += ((nf * sizeof(float) + 255) & ~(size_t)255);
        return p;
    };
    float* hs    = alloc(N * 256);
    float* hv    = alloc(N * 96);
    float* encS  = alloc(N * 256);
    float* encV  = alloc(N * 96);
    float* ecat  = alloc(E * ESCAT);
    float* evb   = alloc(E * 3);
    float* e_vh  = alloc(E * 195);
    float* e_vn  = alloc(E * 65);
    float* e_ms  = alloc(E * 256);
    float* e_ms2 = alloc(E * 256);
    float* e_mv  = alloc(E * 96);
    float* e_mv2 = alloc(E * 96);
    float* dhs   = alloc(N * 256);
    float* dhv   = alloc(N * 96);
    float* deg   = alloc(N);
    float* fsb   = alloc(N * 1024);
    float* fvh   = alloc(N * 192);
    float* fvn   = alloc(N * 64);
    float* fvb   = alloc(N * 192);
    float* fs2   = alloc(N * 256);
    float* fv2   = alloc(N * 96);
    float* xln   = alloc(N * 256);
    float* xg_a  = alloc(N * 768);
    float* xg_b  = alloc(N * 768);
    float* x1    = alloc(N * 512);
    float* x2    = alloc(N * 512);
    float* qb    = alloc(N * 256);
    float* kb    = alloc(N * 256);
    float* vb    = alloc(N * 256);
    float* attn  = alloc(N * N);
    int*   amask = (int*)alloc(E);
    float* outp  = (float*)d_out;

    // ---- launch helpers ----
    auto gemm = [&](const float* A0, const float* B, float* C,
                    long M, int Nc, int K,
                    int a_ld, int a_ks, int a_zs,
                    int b_ks, int b_ns,
                    int c_ld, int c_ns, int c_zs,
                    int nz, const float* bias, int accum, int act,
                    const int* rowidx, const int* msk, const float* A1) {
        dim3 g((Nc + 31) / 32, (unsigned)((M + 127) / 128), nz);
        dim3 b(256);
        bool contig = (a_ks == 1);
        #define GEMM_ARGS A0, A1, rowidx, msk, B, bias, C, (int)M, Nc, K, \
            a_ld, a_ks, a_zs, b_ks, b_ns, c_ld, c_ns, c_zs, accum, act
        if (rowidx) {
            if (msk) {
                if (contig) k_gemm<true,true,true><<<g,b,0,stream>>>(GEMM_ARGS);
                else        k_gemm<true,true,false><<<g,b,0,stream>>>(GEMM_ARGS);
            } else {
                if (contig) k_gemm<true,false,true><<<g,b,0,stream>>>(GEMM_ARGS);
                else        k_gemm<true,false,false><<<g,b,0,stream>>>(GEMM_ARGS);
            }
        } else {
            if (contig)     k_gemm<false,false,true><<<g,b,0,stream>>>(GEMM_ARGS);
            else            k_gemm<false,false,false><<<g,b,0,stream>>>(GEMM_ARGS);
        }
        #undef GEMM_ARGS
    };
    auto fill  = [&](float* p, long n) { k_fill<<<(unsigned)((n+255)/256), 256, 0, stream>>>(p, n); };
    auto vnorm = [&](const float* v, int vld, float* o, int old_, long M, int H) {
        long t = M * H; k_vnorm<<<(unsigned)((t+255)/256), 256, 0, stream>>>(v, o, M, H, vld, old_);
    };
    auto vgate = [&](float* v, int ld, long M, int H) {
        long t = M * H; k_vgate<<<(unsigned)((t+255)/256), 256, 0, stream>>>(v, M, H, ld);
    };
    auto lnrows = [&](float* x, long rows, int ld, int D, const float* g, const float* b) {
        k_layernorm<<<(unsigned)rows, 256, 0, stream>>>(x, g, b, D, ld);
    };
    auto vscale = [&](float* v, long rows, int H, int ld) {
        k_vscale<<<(unsigned)rows, 32, 0, stream>>>(v, H, ld);
    };
    auto resid = [&](float* x, int xld, const float* dh, int dld, const float* dg, long R, int D) {
        long t = R * D; k_residual<<<(unsigned)((t+255)/256), 256, 0, stream>>>(x, xld, dh, dld, dg, R, D);
    };
    auto scat = [&](const float* s, int sld, float* o, int old_, long Ee, int D) {
        long t = Ee * D; k_scatter_add<<<(unsigned)((t+255)/256), 256, 0, stream>>>(s, sld, dstI, o, old_, Ee, D);
    };

    // conv layer (enc: eS/eV/am == null; dec: AR-masked gathers)
    auto conv = [&](const ConvP& p, int ns, const float* eS, const float* eV, const int* am) {
        const int so = ns, h0 = 65;
        // ---- m0 : message GVP over edges ----
        gemm(hv,  p.m0.wh,           e_vh, E, h0, 32, 96,3,1,  h0,1, 195,3,1, 3, nullptr,0,0, srcI, am, eV);
        gemm(evb, p.m0.wh + 32*h0,   e_vh, E, h0,  1,  3,3,1,  h0,1, 195,3,1, 3, nullptr,1,0, nullptr,nullptr,nullptr);
        gemm(hv,  p.m0.wh + 33*h0,   e_vh, E, h0, 32, 96,3,1,  h0,1, 195,3,1, 3, nullptr,1,0, dstI, am, eV);
        vnorm(e_vh, 195, e_vn, 65, E, h0);
        gemm(hs,   p.m0.ws,                         e_ms, E, so, ns, 256,1,0, so,1, 256,1,0, 1, p.m0.bs,0,0, srcI, am, eS);
        gemm(ecat, p.m0.ws + (size_t)ns*so,         e_ms, E, so, 53,  53,1,0, so,1, 256,1,0, 1, nullptr,1,0, nullptr,nullptr,nullptr);
        gemm(hs,   p.m0.ws + (size_t)(ns+53)*so,    e_ms, E, so, ns, 256,1,0, so,1, 256,1,0, 1, nullptr,1,0, dstI, am, eS);
        gemm(e_vn, p.m0.ws + (size_t)(2*ns+53)*so,  e_ms, E, so, h0,  65,1,0, so,1, 256,1,0, 1, nullptr,1,1, nullptr,nullptr,nullptr);
        gemm(e_vh, p.m0.wv, e_mv, E, 32, h0, 195,3,1, 32,1, 96,3,1, 3, nullptr,0,0, nullptr,nullptr,nullptr);
        vgate(e_mv, 96, E, 32);
        // ---- m1 ----
        gemm(e_mv, p.m1.wh, e_vh, E, 32, 32, 96,3,1, 32,1, 195,3,1, 3, nullptr,0,0, nullptr,nullptr,nullptr);
        vnorm(e_vh, 195, e_vn, 65, E, 32);
        gemm(e_ms, p.m1.ws,                  e_ms2, E, so, ns, 256,1,0, so,1, 256,1,0, 1, p.m1.bs,0,0, nullptr,nullptr,nullptr);
        gemm(e_vn, p.m1.ws + (size_t)ns*so,  e_ms2, E, so, 32,  65,1,0, so,1, 256,1,0, 1, nullptr,1,1, nullptr,nullptr,nullptr);
        gemm(e_vh, p.m1.wv, e_mv2, E, 32, 32, 195,3,1, 32,1, 96,3,1, 3, nullptr,0,0, nullptr,nullptr,nullptr);
        vgate(e_mv2, 96, E, 32);
        // ---- m2 (no activations) ----
        gemm(e_mv2, p.m2.wh, e_vh, E, 32, 32, 96,3,1, 32,1, 195,3,1, 3, nullptr,0,0, nullptr,nullptr,nullptr);
        vnorm(e_vh, 195, e_vn, 65, E, 32);
        gemm(e_ms2, p.m2.ws,                 e_ms, E, so, ns, 256,1,0, so,1, 256,1,0, 1, p.m2.bs,0,0, nullptr,nullptr,nullptr);
        gemm(e_vn,  p.m2.ws + (size_t)ns*so, e_ms, E, so, 32,  65,1,0, so,1, 256,1,0, 1, nullptr,1,0, nullptr,nullptr,nullptr);
        gemm(e_vh,  p.m2.wv, e_mv, E, 32, 32, 195,3,1, 32,1, 96,3,1, 3, nullptr,0,0, nullptr,nullptr,nullptr);
        // ---- mean-aggregate + residual + gvp_ln(n0) ----
        fill(dhs, N*256); fill(dhv, N*96); fill(deg, N);
        scat(e_ms, 256, dhs, 256, E, so);
        scat(e_mv,  96, dhv,  96, E, 96);
        k_scatter_deg<<<(unsigned)((E+255)/256), 256, 0, stream>>>(dstI, deg, E);
        resid(hs, 256, dhs, 256, deg, N, so);
        resid(hv,  96, dhv,  96, deg, N, 96);
        lnrows(hs, N, 256, ns, p.n0.g, p.n0.b);
        vscale(hv, N, 32, 96);
        // ---- feed-forward GVPs f0/f1 + residual + gvp_ln(n1) ----
        const int hf = 64, so4 = 4 * ns;
        gemm(hv, p.f0.wh, fvh, N, hf, 32, 96,3,1, hf,1, 192,3,1, 3, nullptr,0,0, nullptr,nullptr,nullptr);
        vnorm(fvh, 192, fvn, 64, N, hf);
        gemm(hs,  p.f0.ws,                   fsb, N, so4, ns, 256,1,0, so4,1, 1024,1,0, 1, p.f0.bs,0,0, nullptr,nullptr,nullptr);
        gemm(fvn, p.f0.ws + (size_t)ns*so4,  fsb, N, so4, hf,  64,1,0, so4,1, 1024,1,0, 1, nullptr,1,1, nullptr,nullptr,nullptr);
        gemm(fvh, p.f0.wv, fvb, N, 64, hf, 192,3,1, 64,1, 192,3,1, 3, nullptr,0,0, nullptr,nullptr,nullptr);
        vgate(fvb, 192, N, 64);
        gemm(fvb, p.f1.wh, fvh, N, hf, 64, 192,3,1, hf,1, 192,3,1, 3, nullptr,0,0, nullptr,nullptr,nullptr);
        vnorm(fvh, 192, fvn, 64, N, hf);
        gemm(fsb, p.f1.ws,                   fs2, N, ns, so4, 1024,1,0, ns,1, 256,1,0, 1, p.f1.bs,0,0, nullptr,nullptr,nullptr);
        gemm(fvn, p.f1.ws + (size_t)so4*ns,  fs2, N, ns, hf,    64,1,0, ns,1, 256,1,0, 1, nullptr,1,0, nullptr,nullptr,nullptr);
        gemm(fvh, p.f1.wv, fv2, N, 32, hf, 192,3,1, 32,1, 96,3,1, 3, nullptr,0,0, nullptr,nullptr,nullptr);
        resid(hs, 256, fs2, 256, nullptr, N, ns);
        resid(hv,  96, fv2,  96, nullptr, N, 96);
        lnrows(hs, N, 256, ns, p.n1.g, p.n1.b);
        vscale(hv, N, 32, 96);
    };

    auto bigru = [&](GruP gf, GruP gb, const float* x, int x_ld, int in_dim, int H,
                     float* out, int out_ld) {
        gemm(x, gf.wih, xg_a, N, 3*H, in_dim, x_ld,1,0, 3*H,1, 3*H,1,0, 1, gf.bih,0,0, nullptr,nullptr,nullptr);
        gemm(x, gb.wih, xg_b, N, 3*H, in_dim, x_ld,1,0, 3*H,1, 3*H,1,0, 1, gb.bih,0,0, nullptr,nullptr,nullptr);
        k_gru_scan<<<1, H, 0, stream>>>(xg_a, gf.whh, gf.bhh, out, (int)N, H, 0, out_ld, 0);
        k_gru_scan<<<1, H, 0, stream>>>(xg_b, gb.whh, gb.bhh, out, (int)N, H, 1, out_ld, H);
    };

    // ======================= forward pass =======================
    // v_ca -> hs[:,0:100], hv[:,0:16]
    {
        GvpP g{P(210), P(211), P(209), P(212)};
        gemm(node_v, g.wh, fvh, N, 16, 3, 9,3,1, 16,1, 48,3,1, 3, nullptr,0,0, nullptr,nullptr,nullptr);
        vnorm(fvh, 48, fvn, 16, N, 16);
        gemm(node_s, g.ws,          hs, N, 100, 6,  6,1,0, 100,1, 256,1,0, 1, g.bs,0,0, nullptr,nullptr,nullptr);
        gemm(fvn,    g.ws + 6*100,  hs, N, 100, 16, 16,1,0, 100,1, 256,1,0, 1, nullptr,1,0, nullptr,nullptr,nullptr);
        gemm(fvh,    g.wv,          hv, N, 16, 16, 48,3,1, 16,1, 96,3,1, 3, nullptr,0,0, nullptr,nullptr,nullptr);
        lnrows(hs, N, 256, 100, P(214), P(213));
        vscale(hv, N, 16, 96);
    }
    // v_c -> hs[:,100:200], hv[:,16:32]
    {
        GvpP g{P(204), P(205), P(203), P(206)};
        gemm(node_no_v, g.wh, fvh, N, 16, 3, 9,3,1, 16,1, 48,3,1, 3, nullptr,0,0, nullptr,nullptr,nullptr);
        vnorm(fvh, 48, fvn, 16, N, 16);
        gemm(node_no_s, g.ws,         hs + 100, N, 100, 6,  6,1,0, 100,1, 256,1,0, 1, g.bs,0,0, nullptr,nullptr,nullptr);
        gemm(fvn,       g.ws + 6*100, hs + 100, N, 100, 16, 16,1,0, 100,1, 256,1,0, 1, nullptr,1,0, nullptr,nullptr,nullptr);
        gemm(fvh,       g.wv,         hv + 48,  N, 16, 16, 48,3,1, 16,1, 96,3,1, 3, nullptr,0,0, nullptr,nullptr,nullptr);
        lnrows(hs + 100, N, 256, 100, P(208), P(207));
        vscale(hv + 48, N, 16, 96);
    }
    // edge GVP -> ecat[:,0:32], evb ; then sequence embedding -> ecat[:,32:53]
    {
        GvpP g{P(100), P(101), P(99), P(102)};
        gemm(edge_v, g.wh, e_vh, E, 1, 1, 3,3,1, 1,1, 3,3,1, 3, nullptr,0,0, nullptr,nullptr,nullptr);
        vnorm(e_vh, 3, e_vn, 1, E, 1);
        gemm(edge_s, g.ws,           ecat, E, 32, 32, 32,1,0, 32,1, ESCAT,1,0, 1, g.bs,0,0, nullptr,nullptr,nullptr);
        gemm(e_vn,   g.ws + 32*32,   ecat, E, 32, 1,   1,1,0, 32,1, ESCAT,1,0, 1, nullptr,1,0, nullptr,nullptr,nullptr);
        gemm(e_vh,   g.wv,           evb,  E, 1, 1, 3,3,1, 1,1, 3,3,1, 3, nullptr,0,0, nullptr,nullptr,nullptr);
        lnrows(ecat, E, ESCAT, 32, P(104), P(103));
        vscale(evb, E, 1, 3);
    }
    k_eemb<<<(unsigned)((E*NAA+255)/256), 256, 0, stream>>>(ecat, srcI, dstI, seq, P(105), E);
    k_armask<<<(unsigned)((E+255)/256), 256, 0, stream>>>(srcI, dstI, amask, E);

    // encoder conv x3
    for (int l = 0; l < 3; ++l) {
        ConvP cp = convp_at(106 + 24 * l);
        conv(cp, NS, nullptr, nullptr, nullptr);
    }

    // encoder RNN: LN -> bigru l0 -> bigru l1
    hipMemcpyAsync(xln, hs, N * 256 * sizeof(float), hipMemcpyDeviceToDevice, stream);
    lnrows(xln, N, 256, NS, P(197), P(196));
    bigru(grup_at(182), grup_at(178), xln, 256, NS, 256, x1, 512);   // l0f / l0b
    bigru(grup_at(190), grup_at(186), x1, 512, 512, 256, x2, 512);   // l1f / l1b

    // self-attention
    gemm(x2, P(1), qb, N, 256, 512, 512,1,0, 256,1, 256,1,0, 1, nullptr,0,0, nullptr,nullptr,nullptr); // wq
    gemm(x2, P(0), kb, N, 256, 512, 512,1,0, 256,1, 256,1,0, 1, nullptr,0,0, nullptr,nullptr,nullptr); // wk
    gemm(x2, P(2), vb, N, 256, 512, 512,1,0, 256,1, 256,1,0, 1, nullptr,0,0, nullptr,nullptr,nullptr); // wv
    gemm(qb, kb, attn, N, (int)N, 256, 256,1,0, 1,256, (int)N,1,0, 1, nullptr,0,0, nullptr,nullptr,nullptr); // Q@K^T
    k_softmax<<<(unsigned)N, 256, 0, stream>>>(attn, (int)N, 0.0625f);
    gemm(attn, vb, hs, N, 256, (int)N, (int)N,1,0, 256,1, 256,1,0, 1, nullptr,0,0, nullptr,nullptr,nullptr); // attn@V
    lnrows(hs, N, 256, 256, P(195), P(194));

    // snapshot encoder state for decoder AR masking
    hipMemcpyAsync(encS, hs, N * 256 * sizeof(float), hipMemcpyDeviceToDevice, stream);
    hipMemcpyAsync(encV, hv, N * 96  * sizeof(float), hipMemcpyDeviceToDevice, stream);

    // decoder x3: AR conv + bigru(128 per dir)
    for (int l = 0; l < 3; ++l) {
        int base = 3 + 32 * l;
        ConvP cp = convp_at(base);
        conv(cp, 256, encS, encV, amask);
        bigru(grup_at(base + 28), grup_at(base + 24), hs, 256, 256, 128, hs, 256);
    }

    // segmentation head: gvp_ln(seg.ln) -> GVP(256+32 -> 1) -> sigmoid
    lnrows(hs, N, 256, 256, P(202), P(201));
    vscale(hv, N, 32, 96);
    gemm(hv, P(199), fvh, N, 32, 32, 96,3,1, 32,1, 192,3,1, 3, nullptr,0,0, nullptr,nullptr,nullptr); // seg wh
    vnorm(fvh, 192, fvn, 32, N, 32);
    gemm(hs,  P(200),       outp, N, 1, 256, 256,1,0, 1,1, 1,1,0, 1, P(198),0,0, nullptr,nullptr,nullptr);
    gemm(fvn, P(200) + 256, outp, N, 1, 32,   32,1,0, 1,1, 1,1,0, 1, nullptr,1,2, nullptr,nullptr,nullptr);
}
